// ScaledDotProductAttention_30588757082942
// MI455X (gfx1250) — matrix operand order repaired and checked by reading, not compiled, not run on hardware
//
#include <hip/hip_runtime.h>
#include <cstdint>
#include <cstddef>

// ---------------------------------------------------------------------------
// MI455X (gfx1250) fused attention:
//   kernel 1: bf16 QKV projection GEMM (V written transposed)
//   kernel 2: 16-query-row attention block, logits tile in 217KB dynamic LDS,
//             v_wmma_f32_16x16x32_bf16 for QK^T and P*V, async Q staging,
//             dual-tile WMMA chains for ILP.
// ---------------------------------------------------------------------------

typedef __attribute__((ext_vector_type(16))) __bf16 v16bf;
typedef __attribute__((ext_vector_type(4)))  __bf16 v4bf;
typedef __attribute__((ext_vector_type(8)))  float  v8f;

constexpr int Bc = 8, Sc = 2048, Dc = 768;
constexpr int BS = Bc * Sc;                         // 16384 flattened rows
constexpr float SCALE = 0.03608439182435161f;       // 1/sqrt(768)

union FragU { v16bf v; uint4 q[2]; };

__device__ __forceinline__ v8f wmma_bf16(const v16bf& a, const v16bf& b, v8f c) {
  // D = A(16x32 bf16) * B(32x16 bf16) + C(16x16 f32)
  return __builtin_amdgcn_wmma_f32_16x16x32_bf16(false, a, false, b, (short)0, c,
                                                 false, false);
}

// ---------------------------------------------------------------------------
// Kernel 1: y = x @ W^T + b   (torch Linear convention), output bf16.
// blockIdx.z selects Q/K/V.  V is stored transposed: VpT[d][b*S+s].
// Workgroup tile: 128 rows x 64 cols; 8 waves, each a 32x32 sub-tile.
// ---------------------------------------------------------------------------
__global__ __launch_bounds__(256)
void proj_kernel(const float* __restrict__ q_in, const float* __restrict__ k_in,
                 const float* __restrict__ v_in,
                 const float* __restrict__ Wq, const float* __restrict__ Wk,
                 const float* __restrict__ Wv,
                 const float* __restrict__ bq, const float* __restrict__ bk,
                 const float* __restrict__ bv,
                 __bf16* __restrict__ Qp, __bf16* __restrict__ Kp,
                 __bf16* __restrict__ VpT)
{
  __shared__ __bf16 Xs[128 * 40];   // 128 rows x 32 k (stride 40 to dodge banks)
  __shared__ __bf16 Ws[64 * 40];    // 64 out-cols x 32 k

  const int tid  = threadIdx.x;
  const int lane = tid & 31;
  const int wv   = tid >> 5;
  const int hi   = (lane >> 4) & 1;  // lane half (K-half selector in this file's k = 16*hi + i order)
  const int n    = lane & 15;
  const int wr   = wv & 3;           // wave row (0..3)
  const int wc   = wv >> 2;          // wave col (0..1)
  const int colwg = blockIdx.x * 64;
  const int rowwg = blockIdx.y * 128;

  const float* X; const float* W; const float* bias; __bf16* out; int tr;
  if (blockIdx.z == 0)      { X = q_in; W = Wq; bias = bq; out = Qp;  tr = 0; }
  else if (blockIdx.z == 1) { X = k_in; W = Wk; bias = bk; out = Kp;  tr = 0; }
  else                      { X = v_in; W = Wv; bias = bv; out = VpT; tr = 1; }

  v8f acc[2][2] = {};

  for (int kb = 0; kb < Dc; kb += 32) {
    // Cooperative load X tile 128x32 fp32 -> bf16 LDS (coalesced float4).
#pragma unroll
    for (int j = 0; j < 4; ++j) {
      int idx = tid + 256 * j;             // 1024 float4 chunks
      int row = idx >> 3, c4 = idx & 7;
      float4 g = *(const float4*)(X + (size_t)(rowwg + row) * Dc + kb + c4 * 4);
      v4bf pk; pk[0] = (__bf16)g.x; pk[1] = (__bf16)g.y;
      pk[2] = (__bf16)g.z; pk[3] = (__bf16)g.w;
      *(v4bf*)(&Xs[row * 40 + c4 * 4]) = pk;
    }
    // Cooperative load W tile 64x32 (row = output col e, contiguous in d).
#pragma unroll
    for (int j = 0; j < 2; ++j) {
      int idx = tid + 256 * j;             // 512 float4 chunks
      int row = idx >> 3, c4 = idx & 7;
      float4 g = *(const float4*)(W + (size_t)(colwg + row) * Dc + kb + c4 * 4);
      v4bf pk; pk[0] = (__bf16)g.x; pk[1] = (__bf16)g.y;
      pk[2] = (__bf16)g.z; pk[3] = (__bf16)g.w;
      *(v4bf*)(&Ws[row * 40 + c4 * 4]) = pk;
    }
    __syncthreads();

    // A fragment: elem i -> K = i + 16*hi  (two 16B LDS chunks)
    FragU a[2], bb[2];
#pragma unroll
    for (int ri = 0; ri < 2; ++ri) {
      int m = 32 * wr + 16 * ri + n;
      a[ri].q[0] = *(const uint4*)(&Xs[m * 40 + hi * 16]);
      a[ri].q[1] = *(const uint4*)(&Xs[m * 40 + hi * 16 + 8]);
    }
    // B fragment: elem i -> K = i + 16*hi  (16 contiguous bf16)
#pragma unroll
    for (int ci = 0; ci < 2; ++ci) {
      int nn = 32 * wc + 16 * ci + n;
      bb[ci].q[0] = *(const uint4*)(&Ws[nn * 40 + hi * 16]);
      bb[ci].q[1] = *(const uint4*)(&Ws[nn * 40 + hi * 16 + 8]);
    }
#pragma unroll
    for (int ri = 0; ri < 2; ++ri)
#pragma unroll
      for (int ci = 0; ci < 2; ++ci)
        acc[ri][ci] = wmma_bf16(a[ri].v, bb[ci].v, acc[ri][ci]);
    __syncthreads();
  }

  // Epilogue: + bias, store bf16 (optionally transposed for V).
#pragma unroll
  for (int ri = 0; ri < 2; ++ri)
#pragma unroll
    for (int ci = 0; ci < 2; ++ci) {
      int colg = colwg + 32 * wc + 16 * ci + n;
      float bv_ = bias[colg];
#pragma unroll
      for (int r = 0; r < 8; ++r) {
        int rowg = rowwg + 32 * wr + 16 * ri + r + 8 * hi;  // C-layout row
        float o = acc[ri][ci][r] + bv_;
        if (tr) out[(size_t)colg * BS + rowg] = (__bf16)o;
        else    out[(size_t)rowg * Dc + colg] = (__bf16)o;
      }
    }
}

// ---------------------------------------------------------------------------
// Kernel 2: attention for one (batch, 16-query block).
// Dynamic LDS: Q tile (bf16), logits tile 16x2048 (f32), P tile (bf16).
// ---------------------------------------------------------------------------
constexpr int LQ = 776;    // Qs stride in bf16 (388 dwords == 4 mod 64 -> bank-friendly)
constexpr int LL = 2052;   // Lg stride in f32
constexpr int LP = 2056;   // Ps stride in bf16 (1028 dwords == 4 mod 64)
constexpr size_t SM_Q = (size_t)16 * LQ * 2;          // 24832 B
constexpr size_t SM_L = (size_t)16 * LL * 4;          // 131328 B
constexpr size_t SM_P = (size_t)16 * LP * 2;          // 65792 B
constexpr size_t SMEM_BYTES = SM_Q + SM_L + SM_P;     // 221952 B (<320KB WGP LDS)

__global__ __launch_bounds__(256)
void attn_kernel(const __bf16* __restrict__ Qp, const __bf16* __restrict__ Kp,
                 const __bf16* __restrict__ VpT,
                 const unsigned char* __restrict__ mask,   // jax bool_: 1 byte
                 float* __restrict__ ctx, float* __restrict__ attn)
{
  extern __shared__ __align__(16) unsigned char smem[];
  __bf16* Qs = (__bf16*)smem;
  float*  Lg = (float*)(smem + SM_Q);
  __bf16* Ps = (__bf16*)(smem + SM_Q + SM_L);

  __shared__ float rowmax_ws[8][16];
  __shared__ float rowmaxF[16];
  __shared__ float rowsum_ws[16][16];
  __shared__ float rowinvF[16];

  const int tid  = threadIdx.x;
  const int lane = tid & 31;
  const int wv   = tid >> 5;        // 8 waves
  const int hi   = (lane >> 4) & 1;
  const int n    = lane & 15;
  const int q0   = blockIdx.x * 16; // query block
  const int b    = blockIdx.y;

  // ---- stage Q tile (16 x 768 bf16) into LDS via async DMA (ASYNCcnt) ----
  {
    const __bf16* qsrc = Qp + (size_t)(b * Sc + q0) * Dc;
#pragma unroll
    for (int j = 0; j < 6; ++j) {
      int idx = tid + 256 * j;            // 1536 chunks of 8 bf16
      int row = idx / 96, c8 = idx % 96;
      // Low 32 bits of the flat shared pointer == LDS byte address (aperture
      // mapping truncates to addr[31:0]); async-to-LDS wants that in a VGPR.
      unsigned lds_addr = (unsigned)(uintptr_t)(&Qs[row * LQ + c8 * 8]);
      const __bf16* gp = qsrc + (size_t)row * Dc + c8 * 8;
      asm volatile("global_load_async_to_lds_b128 %0, %1, off"
                   :: "v"(lds_addr), "v"(gp) : "memory");
    }
    asm volatile("s_wait_asynccnt 0x0" ::: "memory");
  }
  __syncthreads();

  // ---- pass 1: logits = scale * Q K^T into LDS, track per-row max.
  //      Two k-tiles per iteration: shared A fragment, two independent
  //      WMMA accumulation chains (hides WMMA->WMMA RAW latency). ----
  float m8[8];
#pragma unroll
  for (int r = 0; r < 8; ++r) m8[r] = -3.0e38f;

  for (int it = 0; it < 8; ++it) {
    const int jt0 = wv + 16 * it;
    const int jt1 = jt0 + 8;
    const int kj0 = jt0 * 16, kj1 = jt1 * 16;
    const __bf16* kbase0 = Kp + (size_t)(b * Sc + kj0 + n) * Dc;
    const __bf16* kbase1 = Kp + (size_t)(b * Sc + kj1 + n) * Dc;
    if (it + 1 < 8) {  // pull next iteration's K rows toward the WGP
      __builtin_prefetch(kbase0 + (size_t)256 * Dc, 0, 1);
      __builtin_prefetch(kbase1 + (size_t)256 * Dc, 0, 1);
    }
    v8f acc0 = {}, acc1 = {};
    for (int kb = 0; kb < Dc; kb += 32) {
      FragU a, b0, b1;
      a.q[0]  = *(const uint4*)(&Qs[n * LQ + kb + hi * 16]);
      a.q[1]  = *(const uint4*)(&Qs[n * LQ + kb + hi * 16 + 8]);
      b0.q[0] = *(const uint4*)(kbase0 + kb + hi * 16);
      b0.q[1] = *(const uint4*)(kbase0 + kb + hi * 16 + 8);
      b1.q[0] = *(const uint4*)(kbase1 + kb + hi * 16);
      b1.q[1] = *(const uint4*)(kbase1 + kb + hi * 16 + 8);
      acc0 = wmma_bf16(a.v, b0.v, acc0);
      acc1 = wmma_bf16(a.v, b1.v, acc1);
    }
#pragma unroll
    for (int r = 0; r < 8; ++r) {
      float v0 = acc0[r] * SCALE;
      float v1 = acc1[r] * SCALE;
      int lrow = r + 8 * hi;
      Lg[lrow * LL + kj0 + n] = v0;
      Lg[lrow * LL + kj1 + n] = v1;
      m8[r] = fmaxf(m8[r], fmaxf(v0, v1));
    }
  }
  // reduce max within each 16-lane half (rows 0-7 / 8-15), then across waves
#pragma unroll
  for (int r = 0; r < 8; ++r) {
#pragma unroll
    for (int off = 1; off < 16; off <<= 1)
      m8[r] = fmaxf(m8[r], __shfl_xor(m8[r], off, 32));
  }
  if (n == 0) {
#pragma unroll
    for (int r = 0; r < 8; ++r) rowmax_ws[wv][r + 8 * hi] = m8[r];
  }
  __syncthreads();
  if (tid < 16) {
    float m = -3.0e38f;
#pragma unroll
    for (int w2 = 0; w2 < 8; ++w2) m = fmaxf(m, rowmax_ws[w2][tid]);
    rowmaxF[tid] = m;
  }
  __syncthreads();

  // ---- pass 2a: p = exp(l - m), masked -> 0 (max is a safe upper bound);
  //      in-place into Lg; per-thread partial sums ----
  {
    const int row = tid >> 4, seg = tid & 15;
    const float m = rowmaxF[row];
    const unsigned char* mrow = mask + ((size_t)b * Sc + q0 + row) * Sc;
    float s = 0.f;
    for (int i = 0; i < 32; ++i) {
      int col = 4 * (seg + 16 * i);       // interleaved float4 chunks, coalesced
      float4 l4 = *(float4*)(&Lg[row * LL + col]);
      unsigned int mw = *(const unsigned int*)(mrow + col);
      float p0 = (mw & 0x000000ffu) ? 0.f : __expf(l4.x - m);
      float p1 = (mw & 0x0000ff00u) ? 0.f : __expf(l4.y - m);
      float p2 = (mw & 0x00ff0000u) ? 0.f : __expf(l4.z - m);
      float p3 = (mw & 0xff000000u) ? 0.f : __expf(l4.w - m);
      float4 p4 = make_float4(p0, p1, p2, p3);
      *(float4*)(&Lg[row * LL + col]) = p4;
      s += p0 + p1 + p2 + p3;
    }
    rowsum_ws[row][seg] = s;
  }
  __syncthreads();
  if (tid < 16) {
    float s = 0.f;
#pragma unroll
    for (int j = 0; j < 16; ++j) s += rowsum_ws[tid][j];
    rowinvF[tid] = 1.0f / s;
  }
  __syncthreads();

  // ---- pass 2b: normalize; write attn (f32, global) and Ps (bf16, LDS) ----
  {
    const int row = tid >> 4, seg = tid & 15;
    const float inv = rowinvF[row];
    float* arow = attn + ((size_t)b * Sc + q0 + row) * Sc;
    for (int i = 0; i < 32; ++i) {
      int col = 4 * (seg + 16 * i);
      float4 p4 = *(const float4*)(&Lg[row * LL + col]);
      p4.x *= inv; p4.y *= inv; p4.z *= inv; p4.w *= inv;
      *(float4*)(arow + col) = p4;
      v4bf pk; pk[0] = (__bf16)p4.x; pk[1] = (__bf16)p4.y;
      pk[2] = (__bf16)p4.z; pk[3] = (__bf16)p4.w;
      *(v4bf*)(&Ps[row * LP + col]) = pk;
    }
  }
  __syncthreads();

  // ---- pass 3: context = P (16x2048) @ V (2048x768).
  //      Two d-col tiles per iteration: shared A fragment, two chains. ----
  for (int p = 0; p < 3; ++p) {
    const int col0 = (wv + 8 * (2 * p)) * 16;
    const int col1 = (wv + 8 * (2 * p + 1)) * 16;
    const __bf16* vb0 = VpT + (size_t)(col0 + n) * BS + (size_t)b * Sc;
    const __bf16* vb1 = VpT + (size_t)(col1 + n) * BS + (size_t)b * Sc;
    v8f acc0 = {}, acc1 = {};
    for (int kb = 0; kb < Sc; kb += 32) {
      FragU a, b0, b1;
      a.q[0]  = *(const uint4*)(&Ps[n * LP + kb + hi * 16]);
      a.q[1]  = *(const uint4*)(&Ps[n * LP + kb + hi * 16 + 8]);
      b0.q[0] = *(const uint4*)(vb0 + kb + hi * 16);       // V^T: contiguous k
      b0.q[1] = *(const uint4*)(vb0 + kb + hi * 16 + 8);
      b1.q[0] = *(const uint4*)(vb1 + kb + hi * 16);
      b1.q[1] = *(const uint4*)(vb1 + kb + hi * 16 + 8);
      acc0 = wmma_bf16(a.v, b0.v, acc0);
      acc1 = wmma_bf16(a.v, b1.v, acc1);
    }
#pragma unroll
    for (int r = 0; r < 8; ++r) {
      int rowg = q0 + r + 8 * hi;
      ctx[((size_t)b * Sc + rowg) * Dc + col0 + n] = acc0[r];
      ctx[((size_t)b * Sc + rowg) * Dc + col1 + n] = acc1[r];
    }
  }
}

// ---------------------------------------------------------------------------
extern "C" void kernel_launch(void* const* d_in, const int* in_sizes, int n_in,
                              void* d_out, int out_size, void* d_ws, size_t ws_size,
                              hipStream_t stream)
{
  (void)in_sizes; (void)n_in; (void)out_size; (void)ws_size;

  const float* q_in = (const float*)d_in[0];
  const float* k_in = (const float*)d_in[1];
  const float* v_in = (const float*)d_in[2];
  const unsigned char* mask = (const unsigned char*)d_in[3];  // jax bool_ = 1B
  const float* Wq = (const float*)d_in[4];
  const float* bq = (const float*)d_in[5];
  const float* Wk = (const float*)d_in[6];
  const float* bk = (const float*)d_in[7];
  const float* Wv = (const float*)d_in[8];
  const float* bv = (const float*)d_in[9];

  // Workspace: bf16 Qp[BS][D], Kp[BS][D], VpT[D][BS]  (3 * 25.2 MB)
  __bf16* Qp  = (__bf16*)d_ws;
  __bf16* Kp  = Qp + (size_t)BS * Dc;
  __bf16* VpT = Kp + (size_t)BS * Dc;

  // Outputs concatenated in reference return order: context, attn_weights.
  float* ctx  = (float*)d_out;
  float* attn = ctx + (size_t)BS * Dc;

  proj_kernel<<<dim3(Dc / 64, BS / 128, 3), 256, 0, stream>>>(
      q_in, k_in, v_in, Wq, Wk, Wv, bq, bk, bv, Qp, Kp, VpT);

  hipFuncSetAttribute(reinterpret_cast<const void*>(attn_kernel),
                      hipFuncAttributeMaxDynamicSharedMemorySize,
                      (int)SMEM_BYTES);
  attn_kernel<<<dim3(Sc / 16, Bc), 256, SMEM_BYTES, stream>>>(
      Qp, Kp, VpT, mask, ctx, attn);
}
